// social_stgcn_33234456936541
// MI455X (gfx1250) — matrix operand order repaired and checked by reading, not compiled, not run on hardware
//
#include <hip/hip_runtime.h>
#include <hip/hip_bf16.h>
#include <math.h>

// ---------------------------------------------------------------------------
// Social-STGCN forward for MI455X (gfx1250).
// Dominant cost: 11 GEMMs of 2048x2048x2048 (two GCN, 3x3 LSTM gates; the
// H@Th and forget-gate GEMMs are mathematically zero since H=C=0).
// GEMMs run on v_wmma_f32_16x16x32_bf16 with split-bf16 (hi+lo) operands for
// ~f32 accuracy: C += Ahi*Bhi + Ahi*Blo + Alo*Bhi, f32 accumulation.
// Wave tile 64x64 (4x4 WMMA tiles): 48 WMMA per 32 b128 loads = 48 FLOP/B
// from L2 (both operand matrices fit in the 192 MB L2).
// ---------------------------------------------------------------------------

#define NDIM   2048
#define NFELEM (2048u * 2048u)
#define EDGES  131072

typedef __bf16 bf16_t;
typedef __attribute__((ext_vector_type(16))) __bf16 v16bf;
typedef __attribute__((ext_vector_type(8)))  float  v8f;
typedef __attribute__((ext_vector_type(4)))  int    v4i;

static __device__ __forceinline__ bf16_t f2bf(float x) {
  unsigned u = __builtin_bit_cast(unsigned, x);
  unsigned r = (u + 0x7FFFu + ((u >> 16) & 1u)) >> 16;  // round-nearest-even
  unsigned short s = (unsigned short)r;
  return __builtin_bit_cast(bf16_t, s);
}
static __device__ __forceinline__ float bf2f(bf16_t h) {
  unsigned short s = __builtin_bit_cast(unsigned short, h);
  unsigned u = ((unsigned)s) << 16;
  return __builtin_bit_cast(float, u);
}

struct bfrag { v4i a, b; };

// A fragment (16x32 bf16, MxK): lane = 16*half + m. Per documented layout,
// lane holds K = 8*half+[0..7] (VGPR0-3) and K = 16+8*half+[0..7] (VGPR4-7),
// i.e. two contiguous 16B runs of row m.
static __device__ __forceinline__ v16bf fragA(const bf16_t* __restrict__ base,
                                              int row, int k, int half) {
  const bf16_t* r = base + (size_t)row * NDIM + k + 8 * half;
  bfrag f;
  f.a = *(const v4i*)(r);
  f.b = *(const v4i*)(r + 16);
  return __builtin_bit_cast(v16bf, f);
}
// B fragment (32x16 bf16, KxN) with B supplied TRANSPOSED as Bt[N][K]:
// lane = 16*half + n holds K = 8*half+[0..7] and K = 16+8*half+[0..7] of
// column n -> two contiguous 16B runs of Bt row n (same K order as A).
static __device__ __forceinline__ v16bf fragB(const bf16_t* __restrict__ base,
                                              int row, int k, int half) {
  const bf16_t* r = base + (size_t)row * NDIM + k + 8 * half;
  bfrag f;
  f.a = *(const v4i*)(r);
  f.b = *(const v4i*)(r + 16);
  return __builtin_bit_cast(v16bf, f);
}

// ---------------------------------------------------------------------------
// GEMM: C[2048,2048] = A[2048,2048] * B, A and Bt in split bf16.
// 4 waves/block (128 thr), block tile 128x128, wave tile 64x64 (4x4 WMMA).
// ---------------------------------------------------------------------------
__global__ __launch_bounds__(128) void gemm_bf16x3(
    const bf16_t* __restrict__ Ahi, const bf16_t* __restrict__ Alo,
    const bf16_t* __restrict__ Bthi, const bf16_t* __restrict__ Btlo,
    float* __restrict__ C) {
  const int tid  = threadIdx.x;
  const int lane = tid & 31;
  const int w    = tid >> 5;       // 0..3
  const int wm   = w & 1;          // M direction
  const int wn   = w >> 1;         // N direction
  const int m0   = blockIdx.y * 128 + wm * 64;
  const int n0   = blockIdx.x * 128 + wn * 64;
  const int half = lane >> 4;
  const int l16  = lane & 15;

  v8f acc[4][4] = {};
  for (int k0 = 0; k0 < NDIM; k0 += 32) {
    v16bf a_hi[4], a_lo[4], b_hi[4], b_lo[4];
#pragma unroll
    for (int i = 0; i < 4; ++i) {
      a_hi[i] = fragA(Ahi, m0 + 16 * i + l16, k0, half);
      a_lo[i] = fragA(Alo, m0 + 16 * i + l16, k0, half);
      b_hi[i] = fragB(Bthi, n0 + 16 * i + l16, k0, half);
      b_lo[i] = fragB(Btlo, n0 + 16 * i + l16, k0, half);
    }
#pragma unroll
    for (int i = 0; i < 4; ++i)
#pragma unroll
      for (int j = 0; j < 4; ++j) {
        v8f t = acc[i][j];
        t = __builtin_amdgcn_wmma_f32_16x16x32_bf16(false, a_lo[i], false, b_hi[j],
                                                    (short)0, t, false, false);
        t = __builtin_amdgcn_wmma_f32_16x16x32_bf16(false, a_hi[i], false, b_lo[j],
                                                    (short)0, t, false, false);
        t = __builtin_amdgcn_wmma_f32_16x16x32_bf16(false, a_hi[i], false, b_hi[j],
                                                    (short)0, t, false, false);
        acc[i][j] = t;
      }
  }
  // C/D layout: VGPR r, lane L -> m = r + 8*(L/16), n = L%16.
#pragma unroll
  for (int i = 0; i < 4; ++i)
#pragma unroll
    for (int j = 0; j < 4; ++j) {
      const int mbase = m0 + 16 * i + 8 * half;
      const int nn    = n0 + 16 * j + l16;
#pragma unroll
      for (int r = 0; r < 8; ++r)
        C[(size_t)(mbase + r) * NDIM + nn] = acc[i][j][r];
    }
}

// ---------------------------------------------------------------------------
// Elementwise / helper kernels
// ---------------------------------------------------------------------------
__global__ __launch_bounds__(256) void zero_f32(float* __restrict__ p, int n) {
  for (int i = blockIdx.x * blockDim.x + threadIdx.x; i < n;
       i += gridDim.x * blockDim.x)
    p[i] = 0.0f;
}

__global__ __launch_bounds__(256) void cvt_act(const float* __restrict__ x,
                                               bf16_t* __restrict__ hi,
                                               bf16_t* __restrict__ lo) {
  int i = blockIdx.x * 256 + threadIdx.x;
  float v = x[i];
  bf16_t h = f2bf(v);
  hi[i] = h;
  lo[i] = f2bf(v - bf2f(h));
}

// Transpose-convert: W[K=2048][N=2048] f32 -> Wt[N][K] split bf16.
__global__ __launch_bounds__(256) void cvt_wT(const float* __restrict__ W,
                                              bf16_t* __restrict__ hi,
                                              bf16_t* __restrict__ lo) {
  __shared__ float t[32][33];
  const int x0 = blockIdx.x * 32;  // n
  const int y0 = blockIdx.y * 32;  // k
  const int tx = threadIdx.x;
  for (int i = threadIdx.y; i < 32; i += 8)
    t[i][tx] = W[(size_t)(y0 + i) * NDIM + x0 + tx];
  __syncthreads();
  for (int i = threadIdx.y; i < 32; i += 8) {
    float v = t[tx][i];                           // = W[y0+tx][x0+i]
    size_t o = (size_t)(x0 + i) * NDIM + y0 + tx; // Wt[n][k]
    bf16_t h = f2bf(v);
    hi[o] = h;
    lo[o] = f2bf(v - bf2f(h));
  }
}

__global__ __launch_bounds__(256) void count_deg(const int* __restrict__ dst,
                                                 float* __restrict__ deg) {
  int e = blockIdx.x * 256 + threadIdx.x;
  if (e < EDGES)
    __hip_atomic_fetch_add(&deg[dst[e]], 1.0f, __ATOMIC_RELAXED,
                           __HIP_MEMORY_SCOPE_AGENT);
}

__global__ __launch_bounds__(256) void finish_dinv(float* __restrict__ deg) {
  int i = blockIdx.x * 256 + threadIdx.x;
  if (i < NDIM) deg[i] = rsqrtf(deg[i] + 2.0f);
}

// agg[dst] += h[src] * dinv[src]*dinv[dst], one block per edge.
__global__ __launch_bounds__(256) void edge_agg(const int* __restrict__ src,
                                                const int* __restrict__ dst,
                                                const float* __restrict__ dinv,
                                                const float* __restrict__ h,
                                                float* __restrict__ agg) {
  const int e = blockIdx.x;
  const int s = src[e], d = dst[e];
  const float c = dinv[s] * dinv[d];
  const float* hr = h + (size_t)s * NDIM;
  float* ar = agg + (size_t)d * NDIM;
  for (int f = threadIdx.x; f < NDIM; f += 256)
    __hip_atomic_fetch_add(&ar[f], hr[f] * c, __ATOMIC_RELAXED,
                           __HIP_MEMORY_SCOPE_AGENT);
}

// act = relu(agg + 2*dinv^2*h + bias); emit split-bf16 operands for next GEMM.
__global__ __launch_bounds__(256) void gcn_epi(const float* __restrict__ h,
                                               const float* __restrict__ agg,
                                               const float* __restrict__ bias,
                                               const float* __restrict__ dinv,
                                               bf16_t* __restrict__ ahi,
                                               bf16_t* __restrict__ alo) {
  int i = blockIdx.x * 256 + threadIdx.x;
  int r = i >> 11, c = i & 2047;
  float di = dinv[r];
  float v = agg[i] + 2.0f * di * di * h[i] + bias[c];
  v = fmaxf(v, 0.0f);
  bf16_t hh = f2bf(v);
  ahi[i] = hh;
  alo[i] = f2bf(v - bf2f(hh));
}

static __device__ __forceinline__ float sigm(float x) {
  return 1.0f / (1.0f + __expf(-x));
}

// GCLSTM gating with H=C=0: Cn = sigmoid(gi)*tanh(gc), Hn = sigmoid(go+wc2*Cn)*tanh(Cn)
__global__ __launch_bounds__(256) void lstm_gate(
    const float* __restrict__ gi, const float* __restrict__ gc,
    const float* __restrict__ go, const float* __restrict__ bch,
    const float* __restrict__ bb, const float* __restrict__ wc2,
    float* __restrict__ Hout, bf16_t* __restrict__ ahi,
    bf16_t* __restrict__ alo) {
  int i = blockIdx.x * 256 + threadIdx.x;
  int c = i & 2047;
  float iv = sigm(gi[i] + bch[0 * NDIM + c] + bb[0 * NDIM + c]);
  float cv = tanhf(gc[i] + bch[2 * NDIM + c] + bb[2 * NDIM + c]);
  float Cn = iv * cv;
  float ov = sigm(go[i] + bch[3 * NDIM + c] + bb[3 * NDIM + c] + wc2[c] * Cn);
  float Hn = ov * tanhf(Cn);
  Hout[i] = Hn;
  bf16_t hh = f2bf(Hn);
  ahi[i] = hh;
  alo[i] = f2bf(Hn - bf2f(hh));
}

// y = relu(H) @ lin_w^T + lin_b, then log_softmax over groups of 3.
__global__ __launch_bounds__(256) void final_head(const float* __restrict__ H,
                                                  const float* __restrict__ lw,
                                                  const float* __restrict__ lb,
                                                  float* __restrict__ out) {
  __shared__ float red[9][256];
  const int r = blockIdx.x;
  const int t = threadIdx.x;
  float acc[9];
#pragma unroll
  for (int j = 0; j < 9; ++j) acc[j] = 0.0f;
  const float* hr = H + (size_t)r * NDIM;
  for (int k = t; k < NDIM; k += 256) {
    float v = fmaxf(hr[k], 0.0f);
#pragma unroll
    for (int j = 0; j < 9; ++j) acc[j] += v * lw[j * NDIM + k];
  }
#pragma unroll
  for (int j = 0; j < 9; ++j) red[j][t] = acc[j];
  __syncthreads();
  for (int s = 128; s > 0; s >>= 1) {
    if (t < s) {
#pragma unroll
      for (int j = 0; j < 9; ++j) red[j][t] += red[j][t + s];
    }
    __syncthreads();
  }
  if (t == 0) {
    float y[9];
#pragma unroll
    for (int j = 0; j < 9; ++j) y[j] = red[j][0] + lb[j];
#pragma unroll
    for (int g = 0; g < 3; ++g) {
      float m = fmaxf(y[3 * g], fmaxf(y[3 * g + 1], y[3 * g + 2]));
      float se = __expf(y[3 * g] - m) + __expf(y[3 * g + 1] - m) +
                 __expf(y[3 * g + 2] - m);
      float ls = m + __logf(se);
#pragma unroll
      for (int j = 0; j < 3; ++j) out[(size_t)r * 9 + 3 * g + j] = y[3 * g + j] - ls;
    }
  }
}

// ---------------------------------------------------------------------------
extern "C" void kernel_launch(void* const* d_in, const int* in_sizes, int n_in,
                              void* d_out, int out_size, void* d_ws,
                              size_t ws_size, hipStream_t stream) {
  (void)in_sizes; (void)n_in; (void)out_size; (void)ws_size;
  const float* x   = (const float*)d_in[0];
  const int*   ei  = (const int*)d_in[1];
  const float* g1w = (const float*)d_in[3];
  const float* g1b = (const float*)d_in[4];
  const float* g2w = (const float*)d_in[5];
  const float* g2b = (const float*)d_in[6];
  const float* Wx  = (const float*)d_in[7];   // [3,4,F,F]; Th (d_in[8]) is x0
  const float* bch = (const float*)d_in[9];   // [3,4,F]
  const float* wc  = (const float*)d_in[10];  // [3,3,F]
  const float* bb  = (const float*)d_in[11];  // [3,4,F]
  const float* lw  = (const float*)d_in[12];  // [9,F]
  const float* lb  = (const float*)d_in[13];  // [9]
  float* out = (float*)d_out;

  char* w = (char*)d_ws;
  bf16_t* Ahi = (bf16_t*)(w);
  bf16_t* Alo = (bf16_t*)(w + (size_t)NFELEM * 2);
  bf16_t* Whi = (bf16_t*)(w + (size_t)NFELEM * 4);
  bf16_t* Wlo = (bf16_t*)(w + (size_t)NFELEM * 6);
  float* B0   = (float*)(w + (size_t)NFELEM * 8);
  float* B1   = (float*)(w + (size_t)NFELEM * 12);
  float* B2   = (float*)(w + (size_t)NFELEM * 16);
  float* dinv = (float*)(w + (size_t)NFELEM * 20);
  const int* src = ei;
  const int* dst = ei + EDGES;

  const dim3 gblk(128), ggrid(NDIM / 128, NDIM / 128);
  const dim3 tblk(32, 8), tgrid(NDIM / 32, NDIM / 32);
  const int EW_GRID = (int)(NFELEM / 256);  // 16384

  // Degrees -> dinv (shared by both GCN layers).
  zero_f32<<<8, 256, 0, stream>>>(dinv, NDIM);
  count_deg<<<EDGES / 256, 256, 0, stream>>>(dst, dinv);
  finish_dinv<<<NDIM / 256, 256, 0, stream>>>(dinv);

  // x -> split bf16 activation operands.
  cvt_act<<<EW_GRID, 256, 0, stream>>>(x, Ahi, Alo);

  // ---- GCN layer 1 ----
  cvt_wT<<<tgrid, tblk, 0, stream>>>(g1w, Whi, Wlo);
  gemm_bf16x3<<<ggrid, gblk, 0, stream>>>(Ahi, Alo, Whi, Wlo, B0);  // h = x@W
  zero_f32<<<EW_GRID, 256, 0, stream>>>(B1, (int)NFELEM);
  edge_agg<<<EDGES, 256, 0, stream>>>(src, dst, dinv, B0, B1);
  gcn_epi<<<EW_GRID, 256, 0, stream>>>(B0, B1, g1b, dinv, Ahi, Alo);

  // ---- GCN layer 2 ----
  cvt_wT<<<tgrid, tblk, 0, stream>>>(g2w, Whi, Wlo);
  gemm_bf16x3<<<ggrid, gblk, 0, stream>>>(Ahi, Alo, Whi, Wlo, B0);
  zero_f32<<<EW_GRID, 256, 0, stream>>>(B1, (int)NFELEM);
  edge_agg<<<EDGES, 256, 0, stream>>>(src, dst, dinv, B0, B1);
  gcn_epi<<<EW_GRID, 256, 0, stream>>>(B0, B1, g2b, dinv, Ahi, Alo);

  // ---- 3 GCLSTM layers (H=C=0: only gates i, c, o need GEMMs) ----
  for (int l = 0; l < 3; ++l) {
    const size_t wxl = (size_t)l * 4 * NFELEM;
    cvt_wT<<<tgrid, tblk, 0, stream>>>(Wx + wxl + 0 * (size_t)NFELEM, Whi, Wlo);
    gemm_bf16x3<<<ggrid, gblk, 0, stream>>>(Ahi, Alo, Whi, Wlo, B1);  // gate i
    cvt_wT<<<tgrid, tblk, 0, stream>>>(Wx + wxl + 2 * (size_t)NFELEM, Whi, Wlo);
    gemm_bf16x3<<<ggrid, gblk, 0, stream>>>(Ahi, Alo, Whi, Wlo, B2);  // gate c
    cvt_wT<<<tgrid, tblk, 0, stream>>>(Wx + wxl + 3 * (size_t)NFELEM, Whi, Wlo);
    gemm_bf16x3<<<ggrid, gblk, 0, stream>>>(Ahi, Alo, Whi, Wlo, B0);  // gate o
    lstm_gate<<<EW_GRID, 256, 0, stream>>>(B1, B2, B0,
                                           bch + (size_t)l * 4 * NDIM,
                                           bb + (size_t)l * 4 * NDIM,
                                           wc + ((size_t)l * 3 + 2) * NDIM,
                                           B0, Ahi, Alo);
  }

  // ---- Final linear + log_softmax over [3,3] rows ----
  final_head<<<NDIM, 256, 0, stream>>>(B0, lw, lb, out);
}
